// SeparableFcNet_41979010351810
// MI455X (gfx1250) — compile-verified, matrix operand order checked by reading, not run on hardware
//
#include <hip/hip_runtime.h>
#include <hip/hip_bf16.h>

// ---------------------------------------------------------------------------
// Problem constants (from the reference)
// ---------------------------------------------------------------------------
#define T_FRAMES 4
#define PIX_H 64
#define PIX_W 64
#define NP (PIX_H * PIX_W)          // patches per frame = 4096
#define TPN (T_FRAMES * NP)         // 16384 (t,p) sites
#define PS 5
#define PN 75                        // 3*5*5
#define VS 80
#define KNB 14                       // non-local neighbors in inds

// ---------------------------------------------------------------------------
// WMMA types / helpers (CDNA5 gfx1250, wave32)
// ---------------------------------------------------------------------------
typedef __attribute__((ext_vector_type(16))) __bf16 v16bf;
typedef __attribute__((ext_vector_type(8)))  float  v8f;

// native converting cast -> v_cvt_*bf16* (RNE), instead of 4-op bit emulation
__device__ __forceinline__ __bf16 f2bf(float f) { return (__bf16)f; }

// ISA 7.12.2 16-bit fragment layout (16x32 tile, 32 lanes x 16 bf16):
//   owning lane = (row & 15) | ((k & 8) ? 16 : 0)   (row = M for A, N for B)
//   element     = (k & 7) | ((k & 16) >> 1)
__device__ __forceinline__ int frag_lane(int row, int k) {
  return (row & 15) | ((k & 8) ? 16 : 0);
}
__device__ __forceinline__ int frag_elem(int k) {
  return (k & 7) | ((k & 16) >> 1);
}

// ---------------------------------------------------------------------------
// Fused VerHorMat layer:
//   OUT[g,u] = Wh^T[g,h] * ( X[h,v] * Wv[v,u] ) + B[g,u], then optional
//   batchnorm over g, relu, and residual blend  a*blend + (1-a)*val.
// One (t,p) site per block; 4 waves split the 16x16 WMMA tiles.
// All LDS operands are stored pre-swizzled in per-lane fragment layout so
// every WMMA operand is a single aligned 32-byte LDS vector load.
// ---------------------------------------------------------------------------
template <int HIN, int HOUT, int VIN, int VOUT>
__global__ __launch_bounds__(128) void vh_kernel(
    const float* __restrict__ X, long x_tp, int x_rs, int x_ro,
    const float* __restrict__ Wv, const float* __restrict__ Wh,
    const float* __restrict__ Bb,
    const float* __restrict__ bn_g, const float* __restrict__ bn_be,
    const float* __restrict__ bn_m, const float* __restrict__ bn_v,
    int do_relu,
    const float* __restrict__ blend, long bl_tp, int bl_rs, int bl_ro,
    const float* __restrict__ alpha_p,
    float* __restrict__ Out, long o_tp, int o_rs, int o_ro) {
  constexpr int HINP  = ((HIN  + 15) / 16) * 16;   // M pad (ver gemm)
  constexpr int KV    = ((VIN  + 31) / 32) * 32;   // K pad (ver gemm)
  constexpr int KH    = ((HIN  + 31) / 32) * 32;   // K pad (hor gemm)
  constexpr int HOUTP = ((HOUT + 15) / 16) * 16;   // M pad (hor gemm)
  constexpr int VOUTP = ((VOUT + 15) / 16) * 16;   // N pad
  constexpr int NT  = VOUTP / 16;                   // N tiles
  constexpr int MT  = HINP / 16;                    // ver M tiles
  constexpr int MT2 = HOUTP / 16;                   // hor M tiles
  constexpr int CV  = KV / 32;                      // ver K chunks
  constexpr int CH  = KH / 32;                      // hor K chunks

  // fragment-layout LDS: [chunk][tile][lane] -> one v16bf per lane
  __shared__ v16bf sXf [CV * MT  * 32];   // ver A: activations
  __shared__ v16bf sWvf[CV * NT  * 32];   // ver B: ver weights
  __shared__ v16bf sXVf[CH * NT  * 32];   // hor B: ver result
  __shared__ v16bf sWhf[CH * MT2 * 32];   // hor A: hor weights (transposed)

  const int tp   = blockIdx.x;
  const int tid  = threadIdx.x;
  const int lane = tid & 31;
  const int wid  = tid >> 5;

  __builtin_prefetch(Wv, 0, 3);          // global_prefetch_b8: weights -> L2
  __builtin_prefetch(Wh, 0, 3);

  __bf16* pX  = (__bf16*)sXf;
  __bf16* pWv = (__bf16*)sWvf;
  __bf16* pXV = (__bf16*)sXVf;
  __bf16* pWh = (__bf16*)sWhf;

  // ---- stage operands into fragment layout --------------------------------
  const float* Xp = X + (long)tp * x_tp + (long)x_ro * x_rs;
  for (int i = tid; i < HINP * KV; i += 128) {
    const int h = i / KV, k = i - h * KV;
    const float f = (h < HIN && k < VIN) ? Xp[(long)h * x_rs + k] : 0.0f;
    const int idx = (((k >> 5) * MT + (h >> 4)) * 32 + frag_lane(h, k)) * 16 + frag_elem(k);
    pX[idx] = f2bf(f);
  }
  for (int i = tid; i < KV * VOUTP; i += 128) {
    const int k = i / VOUTP, n = i - k * VOUTP;
    const float f = (k < VIN && n < VOUT) ? Wv[k * VOUT + n] : 0.0f;
    const int idx = (((k >> 5) * NT + (n >> 4)) * 32 + frag_lane(n, k)) * 16 + frag_elem(k);
    pWv[idx] = f2bf(f);
  }
  for (int i = tid; i < HOUTP * KH; i += 128) {
    const int g = i / KH, h = i - g * KH;
    const float f = (g < HOUT && h < HIN) ? Wh[h * HOUT + g] : 0.0f;
    const int idx = (((h >> 5) * MT2 + (g >> 4)) * 32 + frag_lane(g, h)) * 16 + frag_elem(h);
    pWh[idx] = f2bf(f);
  }
  // zero the full hor-B buffer; ver gemm overwrites rows [0, HINP) after the
  // barrier, rows [HINP, KH) stay zero (K padding for the hor gemm)
  {
    v16bf z;
#pragma unroll
    for (int e = 0; e < 16; ++e) z[e] = f2bf(0.0f);
    for (int i = tid; i < CH * NT * 32; i += 128) sXVf[i] = z;
  }
  __syncthreads();

  // ---- ver GEMM: (HINP x KV) * (KV x VOUTP) -> sXVf (hor-B layout) --------
  for (int t = wid; t < MT * NT; t += 4) {
    const int mt = t / NT, nt = t - mt * NT;
    v8f acc = {};
#pragma unroll
    for (int c = 0; c < CV; ++c) {
      v16bf a = sXf [(c * MT + mt) * 32 + lane];
      v16bf b = sWvf[(c * NT + nt) * 32 + lane];
      acc = __builtin_amdgcn_wmma_f32_16x16x32_bf16(false, a, false, b,
                                                    (short)0, acc, false, false);
    }
    // scatter D (f32, C/D layout) into the hor-B fragment layout as bf16
    const int n  = nt * 16 + (lane & 15);
    const int mb = mt * 16 + ((lane & 16) ? 8 : 0);
#pragma unroll
    for (int r = 0; r < 8; ++r) {
      const int h = mb + r;   // K index of the hor gemm
      const int idx = (((h >> 5) * NT + nt) * 32 + frag_lane(n, h)) * 16 + frag_elem(h);
      pXV[idx] = f2bf(acc[r]);
    }
  }
  __syncthreads();

  // ---- hor GEMM + fused epilogue ------------------------------------------
  const float alpha = alpha_p ? alpha_p[0] : 0.0f;
  for (int t = wid; t < MT2 * NT; t += 4) {
    const int mt = t / NT, nt = t - mt * NT;
    v8f acc = {};
#pragma unroll
    for (int c = 0; c < CH; ++c) {
      v16bf a = sWhf[(c * MT2 + mt) * 32 + lane];
      v16bf b = sXVf[(c * NT  + nt) * 32 + lane];
      acc = __builtin_amdgcn_wmma_f32_16x16x32_bf16(false, a, false, b,
                                                    (short)0, acc, false, false);
    }
    const int n  = nt * 16 + (lane & 15);
    const int gb = mt * 16 + ((lane & 16) ? 8 : 0);
    if (n < VOUT) {
#pragma unroll
      for (int r = 0; r < 8; ++r) {
        const int g = gb + r;
        if (g < HOUT) {
          float v = acc[r] + Bb[g * VOUT + n];
          if (bn_g) {
            const float s = bn_g[g] * rsqrtf(bn_v[g] + 1e-5f);
            v = (v - bn_m[g]) * s + bn_be[g];
          }
          if (do_relu) v = fmaxf(v, 0.0f);
          if (blend) {
            const float bx = blend[(long)tp * bl_tp + (long)(bl_ro + g) * bl_rs + n];
            v = alpha * bx + (1.0f - alpha) * v;
          }
          Out[(long)tp * o_tp + (long)(o_ro + g) * o_rs + n] = v;
        }
      }
    }
  }
}

// ---------------------------------------------------------------------------
// Aggregation helpers (dnls gather/scatter path)
// ---------------------------------------------------------------------------
__global__ void zero_kernel(float* p, long n) {
  for (long i = (long)blockIdx.x * blockDim.x + threadIdx.x; i < n;
       i += (long)gridDim.x * blockDim.x)
    p[i] = 0.0f;
}

// one thread per (tp, i, j); adds 3 channel values + 1 count
__global__ void agg_scatter(const float* __restrict__ y, const int* __restrict__ inds,
                            const float* __restrict__ wts, float* canvas, float* cnt,
                            int pad, int dil, int H, int W, int n) {
  const int idx = blockIdx.x * blockDim.x + threadIdx.x;
  if (idx >= n) return;
  const int tp = idx / (PS * PS);
  const int ij = idx - tp * (PS * PS);
  const int i = ij / PS, j = ij - (ij / PS) * PS;
  const int* ip = inds + (long)tp * KNB * 3;   // neighbor k=0
  const int ti = ip[0];
  const int hh = ip[1] + pad + i * dil;
  const int ww = ip[2] + pad + j * dil;
  const float invw = wts ? (1.0f / wts[tp]) : 1.0f;
#pragma unroll
  for (int c = 0; c < 3; ++c) {
    const float v = y[(long)tp * PN + c * (PS * PS) + i * PS + j] * invw;
    atomicAdd(&canvas[((long)(ti * 3 + c) * H + hh) * W + ww], v);
  }
  atomicAdd(&cnt[((long)ti * H + hh) * W + ww], 1.0f);
}

__global__ void canvas_div(float* canvas, const float* cnt, int HW, long n) {
  const long idx = (long)blockIdx.x * blockDim.x + threadIdx.x;
  if (idx >= n) return;
  const long t = idx / (3L * HW);
  const long pix = idx % HW;
  canvas[idx] = canvas[idx] / fmaxf(cnt[t * HW + pix], 1.0f);
}

// fixed separable [0.25 0.5 0.25]^2 with reflect padding of 1
__global__ void smooth3(const float* __restrict__ in, float* __restrict__ out,
                        int H, int W, long n) {
  const long idx = (long)blockIdx.x * blockDim.x + threadIdx.x;
  if (idx >= n) return;
  const int x = (int)(idx % W);
  const int y = (int)((idx / W) % H);
  const long tc = idx / ((long)W * H);
  const float* pl = in + tc * H * W;
  const float kw[3] = {0.25f, 0.5f, 0.25f};
  float acc = 0.0f;
#pragma unroll
  for (int dy = 0; dy < 3; ++dy) {
    int yy = y + dy - 1;
    yy = (yy < 0) ? -yy : ((yy >= H) ? (2 * H - 2 - yy) : yy);
#pragma unroll
    for (int dx = 0; dx < 3; ++dx) {
      int xx = x + dx - 1;
      xx = (xx < 0) ? -xx : ((xx >= W) ? (2 * W - 2 - xx) : xx);
      acc += kw[dy] * kw[dx] * pl[yy * W + xx];
    }
  }
  out[idx] = acc;
}

// one thread per (tp, c, i, j) element of the re-extracted patches
__global__ void agg_extract(const float* __restrict__ canvas, const int* __restrict__ inds,
                            const float* __restrict__ wts, float* __restrict__ outp,
                            int pad, int dil, int H, int W, int n) {
  const int idx = blockIdx.x * blockDim.x + threadIdx.x;
  if (idx >= n) return;
  const int tp = idx / PN;
  const int r = idx - tp * PN;
  const int c = r / (PS * PS);
  const int ij = r - c * (PS * PS);
  const int i = ij / PS, j = ij - (ij / PS) * PS;
  const int* ip = inds + (long)tp * KNB * 3;
  const int ti = ip[0];
  const int hh = ip[1] + pad + i * dil;
  const int ww = ip[2] + pad + j * dil;
  float v = canvas[((long)(ti * 3 + c) * H + hh) * W + ww];
  if (wts) v *= wts[tp];
  outp[(long)tp * PN + r] = v;
}

// ---------------------------------------------------------------------------
// Host-side orchestration
// ---------------------------------------------------------------------------
extern "C" void kernel_launch(void* const* d_in, const int* in_sizes, int n_in,
                              void* d_out, int out_size, void* d_ws, size_t ws_size,
                              hipStream_t stream) {
  (void)in_sizes; (void)n_in; (void)out_size; (void)ws_size;

  const float* x0   = (const float*)d_in[0];
  const float* x1   = (const float*)d_in[1];
  const float* wts1 = (const float*)d_in[2];    // (T,P,1,1)
  const int*   in0  = (const int*)d_in[4];
  const int*   in1  = (const int*)d_in[6];
  const float* const* pp = (const float* const*)(d_in + 7);
  // params in insertion order:
  //  s0: 0 v0  1 h0  2 b0  3 a  4 v1  5 h1  6 b1  7 bn_g 8 bn_be 9 bn_m 10 bn_v
  //  s1: 11 v0 12 h0 13 b0 14 a 15 v1 16 h1 17 b1 18..21 bn
  //  22 a0pre_v 23 a0pre_h 24 a0pre_b  25 a0post_v 26 a0post_h 27 a0post_b
  //  28 a1pre_v 29 a1pre_h 30 a1pre_b  31 a1post_v 32 a1post_h 33 a1post_b
  //  34 p2_v2 35 p2_h2 36 p2_b2 37..40 bn2  41 p2_a
  //  42 p2_v3 43 p2_h3 44 p2_b3 45..48 bn3  49 out_v 50 out_h 51 out_b

  // workspace layout (floats)
  float* ws = (float*)d_ws;
  size_t o = 0;
  float* hidden  = ws + o; o += (size_t)TPN * 14 * VS;   // part1 layer-1 result
  float* zbuf    = ws + o; o += (size_t)TPN * 56 * VS;   // concat z; reused as wf
  float* wbuf    = ws + o; o += (size_t)TPN * 56 * VS;   // p2 layer-1 result
  float* ypre    = ws + o; o += (size_t)TPN * VS;        // (t,p,75) pre-agg
  float* yagg    = ws + o; o += (size_t)TPN * VS;        // (t,p,75) post-agg
  float* canvas  = ws + o; o += (size_t)T_FRAMES * 3 * 72 * 72;
  float* canvas2 = ws + o; o += (size_t)T_FRAMES * 3 * 72 * 72;
  float* cnt     = ws + o; o += (size_t)T_FRAMES * 72 * 72;

  const dim3 G(TPN), B(128);
  const long Z_TP = 56L * VS;

  // ---- part1, branch s0: z rows 0..13 -------------------------------------
  vh_kernel<14, 14, 75, 80><<<G, B, 0, stream>>>(
      x0, 14L * PN, PN, 0, pp[0], pp[1], pp[2],
      nullptr, nullptr, nullptr, nullptr, 1,
      nullptr, 0, 0, 0, nullptr, hidden, 14L * VS, VS, 0);
  vh_kernel<14, 14, 80, 80><<<G, B, 0, stream>>>(
      hidden, 14L * VS, VS, 0, pp[4], pp[5], pp[6],
      pp[7], pp[8], pp[9], pp[10], 1,
      hidden, 14L * VS, VS, 0, pp[3], zbuf, Z_TP, VS, 0);

  // ---- part1, branch s1: z rows 14..27 ------------------------------------
  vh_kernel<14, 14, 75, 80><<<G, B, 0, stream>>>(
      x1, 14L * PN, PN, 0, pp[11], pp[12], pp[13],
      nullptr, nullptr, nullptr, nullptr, 1,
      nullptr, 0, 0, 0, nullptr, hidden, 14L * VS, VS, 0);
  vh_kernel<14, 14, 80, 80><<<G, B, 0, stream>>>(
      hidden, 14L * VS, VS, 0, pp[15], pp[16], pp[17],
      pp[18], pp[19], pp[20], pp[21], 1,
      hidden, 14L * VS, VS, 0, pp[14], zbuf, Z_TP, VS, 14);

  // ---- branch 0 aggregation: z rows 28..41 --------------------------------
  vh_kernel<14, 1, 80, 75><<<G, B, 0, stream>>>(
      zbuf, Z_TP, VS, 0, pp[22], pp[23], pp[24],
      nullptr, nullptr, nullptr, nullptr, 0,
      nullptr, 0, 0, 0, nullptr, ypre, (long)PN, PN, 0);
  {
    const long nc = (long)T_FRAMES * 3 * 72 * 72, ncnt = (long)T_FRAMES * 72 * 72;
    zero_kernel<<<256, 256, 0, stream>>>(canvas, nc);
    zero_kernel<<<64, 256, 0, stream>>>(cnt, ncnt);
    const int ns = TPN * PS * PS;
    agg_scatter<<<(ns + 255) / 256, 256, 0, stream>>>(ypre, in0, nullptr, canvas, cnt, 2, 1, 72, 72, ns);
    canvas_div<<<(int)((nc + 255) / 256), 256, 0, stream>>>(canvas, cnt, 72 * 72, nc);
    const int ne = TPN * PN;
    agg_extract<<<(ne + 255) / 256, 256, 0, stream>>>(canvas, in0, nullptr, yagg, 2, 1, 72, 72, ne);
  }
  vh_kernel<1, 14, 75, 80><<<G, B, 0, stream>>>(
      yagg, (long)PN, PN, 0, pp[25], pp[26], pp[27],
      nullptr, nullptr, nullptr, nullptr, 1,
      nullptr, 0, 0, 0, nullptr, zbuf, Z_TP, VS, 28);

  // ---- branch 1 aggregation (weighted + smoothing): z rows 42..55 ---------
  vh_kernel<14, 1, 80, 75><<<G, B, 0, stream>>>(
      zbuf, Z_TP, VS, 14, pp[28], pp[29], pp[30],
      nullptr, nullptr, nullptr, nullptr, 0,
      nullptr, 0, 0, 0, nullptr, ypre, (long)PN, PN, 0);
  {
    const long nc = (long)T_FRAMES * 3 * PIX_H * PIX_W, ncnt = (long)T_FRAMES * PIX_H * PIX_W;
    zero_kernel<<<256, 256, 0, stream>>>(canvas, nc);
    zero_kernel<<<64, 256, 0, stream>>>(cnt, ncnt);
    const int ns = TPN * PS * PS;
    agg_scatter<<<(ns + 255) / 256, 256, 0, stream>>>(ypre, in1, wts1, canvas, cnt, 4, 2, PIX_H, PIX_W, ns);
    canvas_div<<<(int)((nc + 255) / 256), 256, 0, stream>>>(canvas, cnt, PIX_H * PIX_W, nc);
    smooth3<<<(int)((nc + 255) / 256), 256, 0, stream>>>(canvas, canvas2, PIX_H, PIX_W, nc);
    const int ne = TPN * PN;
    agg_extract<<<(ne + 255) / 256, 256, 0, stream>>>(canvas2, in1, wts1, yagg, 4, 2, PIX_H, PIX_W, ne);
  }
  vh_kernel<1, 14, 75, 80><<<G, B, 0, stream>>>(
      yagg, (long)PN, PN, 0, pp[31], pp[32], pp[33],
      nullptr, nullptr, nullptr, nullptr, 1,
      nullptr, 0, 0, 0, nullptr, zbuf, Z_TP, VS, 42);

  // ---- part2 --------------------------------------------------------------
  vh_kernel<56, 56, 80, 80><<<G, B, 0, stream>>>(
      zbuf, Z_TP, VS, 0, pp[34], pp[35], pp[36],
      pp[37], pp[38], pp[39], pp[40], 1,
      nullptr, 0, 0, 0, nullptr, wbuf, Z_TP, VS, 0);
  vh_kernel<56, 56, 80, 80><<<G, B, 0, stream>>>(
      wbuf, Z_TP, VS, 0, pp[42], pp[43], pp[44],
      pp[45], pp[46], pp[47], pp[48], 1,
      wbuf, Z_TP, VS, 0, pp[41], zbuf, Z_TP, VS, 0);   // zbuf reused as w_final

  // ---- output projection --------------------------------------------------
  vh_kernel<56, 1, 80, 75><<<G, B, 0, stream>>>(
      zbuf, Z_TP, VS, 0, pp[49], pp[50], pp[51],
      nullptr, nullptr, nullptr, nullptr, 0,
      nullptr, 0, 0, 0, nullptr, (float*)d_out, (long)PN, PN, 0);
}
